// InvariantPointAttention_78640851189924
// MI455X (gfx1250) — matrix operand order repaired and checked by reading, not compiled, not run on hardware
//
#include <hip/hip_runtime.h>
#include <hip/hip_bf16.h>
#include <math.h>

// ---------------- problem constants ----------------
#define NRES   768
#define C_S    384
#define C_Z    128
#define C_IPA  16
#define NH     12
#define NQPTS  4
#define NVPTS  8

#define NORMSIZE    0.14433756729740643f   // sqrt(1/48)
#define NORMSIZE_B  0.5773502691896258f    // sqrt(1/3)
#define NORMSIZE_HW 0.13608276348795434f   // sqrt(1/54)

typedef __attribute__((ext_vector_type(16))) __bf16 bf16x16;
typedef __attribute__((ext_vector_type(8)))  float  v8f;

// ---------------------------------------------------------------------------
// Generic batched GEMM: C[b] = A[b](MxK) * B[b](KxN) + bias.
// fp32 operands -> bf16 in-register (v_cvt_pk_bf16_f32), fp32 accumulate via
// v_wmma_f32_16x16x32_bf16. One wave owns a 16x(16*NT) C tile, A fragment
// reused across the NT column tiles. K must be a multiple of 32.
// BTR=true: B is stored transposed (Ncol x K, ldb = col stride = K) so the
// B fragment is 4 x b128 contiguous loads per lane, same as A.
// BTR=false: B is K x Ncol row-major, per-lane strided b32 loads (used only
// where transposing would cost more traffic than it saves, i.e. B = z).
// ---------------------------------------------------------------------------
template<int NT, bool MCHK, bool NCHK, bool BTR>
__global__ void wmma_gemm(const float* __restrict__ A, const float* __restrict__ B,
                          float* __restrict__ C, const float* __restrict__ bias,
                          int M, int K, int Ncol,
                          int lda, int ldb, int ldc,
                          long batchA, long batchB, long batchC) {
    const int lane = threadIdx.x;                        // 0..31 (wave32)
    const int row0 = (blockIdx.y * blockDim.y + threadIdx.y) * 16;
    const int col0 = blockIdx.x * (16 * NT);
    if (row0 >= M || col0 >= Ncol) return;               // wave-uniform exit

    A += (long)blockIdx.z * batchA;
    B += (long)blockIdx.z * batchB;
    C += (long)blockIdx.z * batchC;

    const int  l15  = lane & 15;
    const int  mA   = row0 + l15;
    const bool aOk  = !MCHK || (mA < M);
    const float aMul = aOk ? 1.0f : 0.0f;
    // 32-bit per-lane element offset into A (clamped to a valid row)
    const unsigned aOff = (unsigned)((aOk ? mA : row0) * lda + ((lane < 16) ? 0 : 16));

    const int kHiB = (lane < 16) ? 0 : 16;
    int nBi[NT]; float bMul[NT]; unsigned bOff[NT];
    #pragma unroll
    for (int t = 0; t < NT; ++t) {
        int n   = col0 + 16 * t + l15;
        nBi[t]  = n;
        bool ok = !NCHK || (n < Ncol);
        bMul[t] = ok ? 1.0f : 0.0f;
        int nc  = ok ? n : col0;                         // clamped col
        bOff[t] = BTR ? (unsigned)(nc * ldb + kHiB)      // transposed: col-major base
                      : (unsigned)(kHiB * ldb + nc);     // row-major
    }

    v8f acc[NT] = {};
    for (int k0 = 0; k0 < K; k0 += 32) {
        // ---- load phase ----
        const float* Ak = A + (size_t)(aOff + (unsigned)k0);
        const float4 a0 = *(const float4*)(Ak);
        const float4 a1 = *(const float4*)(Ak + 4);
        const float4 a2 = *(const float4*)(Ak + 8);
        const float4 a3 = *(const float4*)(Ak + 12);

        float4 bq[NT][4];                                // BTR path: b128 loads
        float  bv[NT][16];                               // row-major path
        if (BTR) {
            #pragma unroll
            for (int t = 0; t < NT; ++t) {
                const float* Bl = B + (size_t)(bOff[t] + (unsigned)k0);
                bq[t][0] = *(const float4*)(Bl);
                bq[t][1] = *(const float4*)(Bl + 4);
                bq[t][2] = *(const float4*)(Bl + 8);
                bq[t][3] = *(const float4*)(Bl + 12);
            }
        } else {
            const float* Bk = B + (size_t)k0 * (unsigned)ldb;
            #pragma unroll
            for (int t = 0; t < NT; ++t)
                #pragma unroll
                for (int e = 0; e < 16; ++e)
                    bv[t][e] = Bk[(size_t)(bOff[t] + (unsigned)(e * ldb))];
        }

        // ---- convert phase ----
        bf16x16 af;
        if (MCHK) {
            af[0]=(__bf16)(a0.x*aMul); af[1]=(__bf16)(a0.y*aMul);
            af[2]=(__bf16)(a0.z*aMul); af[3]=(__bf16)(a0.w*aMul);
            af[4]=(__bf16)(a1.x*aMul); af[5]=(__bf16)(a1.y*aMul);
            af[6]=(__bf16)(a1.z*aMul); af[7]=(__bf16)(a1.w*aMul);
            af[8]=(__bf16)(a2.x*aMul); af[9]=(__bf16)(a2.y*aMul);
            af[10]=(__bf16)(a2.z*aMul); af[11]=(__bf16)(a2.w*aMul);
            af[12]=(__bf16)(a3.x*aMul); af[13]=(__bf16)(a3.y*aMul);
            af[14]=(__bf16)(a3.z*aMul); af[15]=(__bf16)(a3.w*aMul);
        } else {
            af[0]=(__bf16)a0.x; af[1]=(__bf16)a0.y; af[2]=(__bf16)a0.z; af[3]=(__bf16)a0.w;
            af[4]=(__bf16)a1.x; af[5]=(__bf16)a1.y; af[6]=(__bf16)a1.z; af[7]=(__bf16)a1.w;
            af[8]=(__bf16)a2.x; af[9]=(__bf16)a2.y; af[10]=(__bf16)a2.z; af[11]=(__bf16)a2.w;
            af[12]=(__bf16)a3.x; af[13]=(__bf16)a3.y; af[14]=(__bf16)a3.z; af[15]=(__bf16)a3.w;
        }
        #pragma unroll
        for (int t = 0; t < NT; ++t) {
            bf16x16 bfv;
            if (BTR) {
                const float m = bMul[t];
                #pragma unroll
                for (int q = 0; q < 4; ++q) {
                    float4 x = bq[t][q];
                    if (NCHK) { x.x*=m; x.y*=m; x.z*=m; x.w*=m; }
                    bfv[q*4+0]=(__bf16)x.x; bfv[q*4+1]=(__bf16)x.y;
                    bfv[q*4+2]=(__bf16)x.z; bfv[q*4+3]=(__bf16)x.w;
                }
            } else {
                #pragma unroll
                for (int e = 0; e < 16; ++e)
                    bfv[e] = NCHK ? (__bf16)(bv[t][e] * bMul[t]) : (__bf16)bv[t][e];
            }
            acc[t] = __builtin_amdgcn_wmma_f32_16x16x32_bf16(
                         false, af, false, bfv, (short)0, acc[t], false, false);
        }
    }

    const int mBase = (lane < 16) ? 0 : 8;               // D: VGPR v -> row v (+8 hi half)
    #pragma unroll
    for (int t = 0; t < NT; ++t) {
        const bool colOk = !NCHK || (nBi[t] < Ncol);     // folds when !NCHK
        if (colOk) {
            float bvadd = bias ? bias[nBi[t]] : 0.0f;
            #pragma unroll
            for (int v = 0; v < 8; ++v) {
                int m = row0 + mBase + v;
                const bool rowOk = !MCHK || (m < M);     // folds when !MCHK
                if (rowOk)
                    C[(size_t)((unsigned)m * (unsigned)ldc + (unsigned)nBi[t])]
                        = acc[t][v] + bvadd;
            }
        }
    }
}

// ---------------------------------------------------------------------------
// transpose: in (K x N row-major) -> outT (N x K row-major)
// ---------------------------------------------------------------------------
__global__ void transpose_kernel(const float* __restrict__ in, float* __restrict__ outT,
                                 int K, int N) {
    int idx = blockIdx.x * blockDim.x + threadIdx.x;
    if (idx < K * N) {
        int k = idx / N, n = idx % N;
        outT[(long)n * K + k] = in[idx];
    }
}

// ---------------------------------------------------------------------------
// pad+transpose Wb (128x12) -> WbpT (16x128, rows 12..15 zero); bb -> bbp (16)
// ---------------------------------------------------------------------------
__global__ void pad_wb_kernel(const float* __restrict__ Wb, const float* __restrict__ bb,
                              float* __restrict__ WbpT, float* __restrict__ bbp) {
    int t = blockIdx.x * blockDim.x + threadIdx.x;       // 0 .. 16*128-1
    if (t < 16 * C_Z) {
        int n = t / C_Z, k = t % C_Z;
        WbpT[t] = (n < NH) ? Wb[k * NH + n] : 0.0f;
    }
    if (t < 16) bbp[t] = (t < NH) ? bb[t] : 0.0f;
}

// ---------------------------------------------------------------------------
// prep: rotate Q3 / KV3 points into global frame; assemble transposed
// per-head B-matrix VcatT[h][col(64)][j]: col 0:16=V, 16:40=V3, 40:64=0.
// ---------------------------------------------------------------------------
__global__ void prep_kernel(const float* __restrict__ Q3raw, const float* __restrict__ KV3raw,
                            const float* __restrict__ KVf,
                            const float* __restrict__ rots, const float* __restrict__ trans,
                            float* __restrict__ Q3buf, float* __restrict__ K3buf,
                            float* __restrict__ VcatT) {
    const int n = blockIdx.x;
    const int t = threadIdx.x;                           // 0..191
    __shared__ float R[9], T[3];
    if (t < 9) R[t] = rots[n * 9 + t];
    if (t < 3) T[t] = trans[n * 3 + t];
    __syncthreads();

    if (t < NQPTS * NH) {                                // 48 query points
        float r0 = Q3raw[n * 144 + t];
        float r1 = Q3raw[n * 144 + 48 + t];
        float r2 = Q3raw[n * 144 + 96 + t];
        #pragma unroll
        for (int e = 0; e < 3; ++e)
            Q3buf[(long)n * 144 + t * 3 + e] = R[e*3+0]*r0 + R[e*3+1]*r1 + R[e*3+2]*r2 + T[e];
    }
    if (t < (NQPTS + NVPTS) * NH) {                      // 144 key/value points
        float r0 = KV3raw[n * 432 + t];
        float r1 = KV3raw[n * 432 + 144 + t];
        float r2 = KV3raw[n * 432 + 288 + t];
        int h = t / 12, c = t % 12;
        #pragma unroll
        for (int e = 0; e < 3; ++e) {
            float o = R[e*3+0]*r0 + R[e*3+1]*r1 + R[e*3+2]*r2 + T[e];
            if (c < NQPTS)
                K3buf[(long)n * 144 + (h * NQPTS + c) * 3 + e] = o;
            else
                VcatT[((long)h * 64 + 16 + (c - NQPTS) * 3 + e) * NRES + n] = o;
        }
    }
    {                                                    // scalar V into cols 0..15
        int h = t / 16, c = t % 16;
        VcatT[((long)h * 64 + c) * NRES + n] = KVf[(long)n * 384 + h * 32 + 16 + c];
    }
    for (int p = t; p < NH * 24; p += 192) {             // zero pad cols 40..63
        int h = p / 24, c = 40 + p % 24;
        VcatT[((long)h * 64 + c) * NRES + n] = 0.0f;
    }
}

// ---------------------------------------------------------------------------
// attention logits + softmax over j, one block per (i, h); 256 threads x 3 j's.
// ---------------------------------------------------------------------------
__global__ void attn_softmax(const float* __restrict__ Qf, const float* __restrict__ KVf,
                             const float* __restrict__ Q3buf, const float* __restrict__ K3buf,
                             const float* __restrict__ Bmat, const float* __restrict__ mask,
                             const float* __restrict__ hw, float* __restrict__ ATf) {
    const int i = blockIdx.x, h = blockIdx.y;
    const int t = threadIdx.x;                           // 0..255
    __shared__ float red[256];
    __shared__ float qv[16], q3[12];
    if (t < 16) qv[t] = Qf[(long)i * 192 + h * 16 + t];
    if (t < 12) q3[t] = Q3buf[(long)i * 144 + h * 12 + t];
    __syncthreads();

    const float hwsp = logf(1.0f + expf(hw[h])) * NORMSIZE_HW;
    const float mi   = mask[i];

    float logit[3];
    float maxv = -3.0e38f;
    #pragma unroll
    for (int r = 0; r < 3; ++r) {
        int j = t + r * 256;
        const float* kp = &KVf[(long)j * 384 + h * 32];
        float dot = 0.0f;
        #pragma unroll
        for (int c = 0; c < 16; ++c) dot += qv[c] * kp[c];
        const float* k3 = &K3buf[(long)j * 144 + h * 12];
        float d2 = 0.0f;
        #pragma unroll
        for (int e = 0; e < 12; ++e) { float d = q3[e] - k3[e]; d2 += d * d; }
        float bterm = Bmat[((long)i * NRES + j) * 16 + h];
        float l = dot * NORMSIZE + bterm * NORMSIZE_B - 0.5f * hwsp * d2
                + (mi * mask[j] - 1.0f) * 1.0e9f;
        logit[r] = l;
        maxv = fmaxf(maxv, l);
    }
    red[t] = maxv; __syncthreads();
    for (int s = 128; s >= 1; s >>= 1) {
        if (t < s) red[t] = fmaxf(red[t], red[t + s]);
        __syncthreads();
    }
    maxv = red[0]; __syncthreads();

    float sum = 0.0f;
    #pragma unroll
    for (int r = 0; r < 3; ++r) { logit[r] = expf(logit[r] - maxv); sum += logit[r]; }
    red[t] = sum; __syncthreads();
    for (int s = 128; s >= 1; s >>= 1) {
        if (t < s) red[t] += red[t + s];
        __syncthreads();
    }
    const float inv = 1.0f / red[0];
    #pragma unroll
    for (int r = 0; r < 3; ++r)
        ATf[(long)h * NRES * NRES + (long)i * NRES + (t + r * 256)] = logit[r] * inv;
}

// ---------------------------------------------------------------------------
// epilogue: scatter O, inverse-rotate O3, norms, into O_final (i x 2112).
// ---------------------------------------------------------------------------
__global__ void epilogue_kernel(const float* __restrict__ Ocat,
                                const float* __restrict__ rots, const float* __restrict__ trans,
                                float* __restrict__ Ofinal) {
    const int i = blockIdx.x;
    const int t = threadIdx.x;                           // 0..191
    {                                                    // scalar O: cols 0..191
        int h = t / 16, c = t % 16;
        Ofinal[(long)i * 2112 + h * 16 + c] = Ocat[((long)h * NRES + i) * 64 + c];
    }
    if (t < NH * NVPTS) {                                // 96 point outputs
        int h = t / 8, c = t % 8;
        float w[3], o[3];
        #pragma unroll
        for (int d = 0; d < 3; ++d)
            w[d] = Ocat[((long)h * NRES + i) * 64 + 16 + c * 3 + d] - trans[i * 3 + d];
        #pragma unroll
        for (int e = 0; e < 3; ++e)                      // R^T * w
            o[e] = rots[i*9 + 0*3 + e] * w[0] + rots[i*9 + 1*3 + e] * w[1]
                 + rots[i*9 + 2*3 + e] * w[2];
        float nrm = sqrtf(o[0]*o[0] + o[1]*o[1] + o[2]*o[2] + 1.0e-20f);
        int k = h * NVPTS + c;
        Ofinal[(long)i * 2112 + 192 + k] = o[0];
        Ofinal[(long)i * 2112 + 288 + k] = o[1];
        Ofinal[(long)i * 2112 + 384 + k] = o[2];
        Ofinal[(long)i * 2112 + 480 + k] = nrm;
    }
}

// ---------------------------------------------------------------------------
static inline dim3 gemm_grid(int M, int Ncol, int nt, int batch) {
    return dim3((Ncol + 16 * nt - 1) / (16 * nt), (M + 63) / 64, batch);
}
static inline void transpose(const float* in, float* outT, int K, int N, hipStream_t st) {
    transpose_kernel<<<dim3((K * N + 255) / 256), dim3(256), 0, st>>>(in, outT, K, N);
}

extern "C" void kernel_launch(void* const* d_in, const int* in_sizes, int n_in,
                              void* d_out, int out_size, void* d_ws, size_t ws_size,
                              hipStream_t stream) {
    const float* s    = (const float*)d_in[0];
    const float* z    = (const float*)d_in[1];
    const float* rots = (const float*)d_in[2];
    const float* trn  = (const float*)d_in[3];
    const float* mask = (const float*)d_in[4];
    const float* Wq   = (const float*)d_in[5];
    const float* bq   = (const float*)d_in[6];
    const float* Wkv  = (const float*)d_in[7];
    const float* bkv  = (const float*)d_in[8];
    const float* Wq3  = (const float*)d_in[9];
    const float* bq3  = (const float*)d_in[10];
    const float* Wkv3 = (const float*)d_in[11];
    const float* bkv3 = (const float*)d_in[12];
    const float* Wb   = (const float*)d_in[13];
    const float* bb   = (const float*)d_in[14];
    const float* hw   = (const float*)d_in[15];
    const float* Wo   = (const float*)d_in[16];
    const float* bo   = (const float*)d_in[17];
    float* out = (float*)d_out;

    // workspace carve-up (fp32 elements)
    float* w = (float*)d_ws;
    float* Qf     = w;               w += (long)NRES * 192;        // s@Wq
    float* KVf    = w;               w += (long)NRES * 384;        // s@Wkv
    float* Q3raw  = w;               w += (long)NRES * 144;
    float* KV3raw = w;               w += (long)NRES * 432;
    float* Q3buf  = w;               w += (long)NRES * 144;        // rotated q points
    float* K3buf  = w;               w += (long)NRES * 144;        // rotated k points
    float* VcatT  = w;               w += (long)NH * 64 * NRES;    // [V|V3|0]^T per head
    float* Ocat   = w;               w += (long)NH * NRES * 64;    // AT@[V|V3]
    float* Ofinal = w;               w += (long)NRES * 2112;       // concat features
    float* WqT    = w;               w += (long)192 * C_S;         // transposed weights
    float* WkvT   = w;               w += (long)384 * C_S;
    float* Wq3T   = w;               w += (long)144 * C_S;
    float* Wkv3T  = w;               w += (long)432 * C_S;
    float* WoT    = w;               w += (long)C_S * 2112;
    float* WbpT   = w;               w += (long)16 * C_Z;          // padded+T Wb
    float* bbp    = w;               w += 16;                      // padded bb
    float* Bmat   = w;               w += (long)NRES * NRES * 16;  // z@Wbp
    float* ATf    = w;               w += (long)NH * NRES * NRES;  // softmax probs

    const long NN = (long)NRES * NRES;
    const dim3 blk(32, 4, 1);

    // 0: transpose weights (one-time, ~5 MB total) + pad Wb/bb
    transpose(Wq,   WqT,   C_S, 192, stream);
    transpose(Wkv,  WkvT,  C_S, 384, stream);
    transpose(Wq3,  Wq3T,  C_S, 144, stream);
    transpose(Wkv3, Wkv3T, C_S, 432, stream);
    transpose(Wo,   WoT,   2112, C_S, stream);
    pad_wb_kernel<<<dim3(8), dim3(256), 0, stream>>>(Wb, bb, WbpT, bbp);

    // 1-4: projections from s (WMMA, transposed-B path: all-b128 loads)
    wmma_gemm<2,false,false,true><<<gemm_grid(NRES,192,2,1), blk, 0, stream>>>(
        s, WqT, Qf, bq, NRES, C_S, 192, C_S, C_S, 192, 0, 0, 0);
    wmma_gemm<2,false,false,true><<<gemm_grid(NRES,384,2,1), blk, 0, stream>>>(
        s, WkvT, KVf, bkv, NRES, C_S, 384, C_S, C_S, 384, 0, 0, 0);
    wmma_gemm<2,false,true,true><<<gemm_grid(NRES,144,2,1), blk, 0, stream>>>(
        s, Wq3T, Q3raw, bq3, NRES, C_S, 144, C_S, C_S, 144, 0, 0, 0);
    wmma_gemm<2,false,true,true><<<gemm_grid(NRES,432,2,1), blk, 0, stream>>>(
        s, Wkv3T, KV3raw, bkv3, NRES, C_S, 432, C_S, C_S, 432, 0, 0, 0);

    // 5: pair bias Bmat = z @ Wbp + bbp  (big streaming GEMM, mask-free)
    wmma_gemm<1,false,false,true><<<gemm_grid(NRES*NRES,16,1,1), blk, 0, stream>>>(
        z, WbpT, Bmat, bbp, NRES * NRES, C_Z, 16, C_Z, C_Z, 16, 0, 0, 0);

    // 6: rotate points, build VcatT
    prep_kernel<<<dim3(NRES), dim3(192), 0, stream>>>(Q3raw, KV3raw, KVf, rots, trn,
                                                      Q3buf, K3buf, VcatT);

    // 7: fused logits + softmax -> AT
    attn_softmax<<<dim3(NRES, NH), dim3(256), 0, stream>>>(Qf, KVf, Q3buf, K3buf,
                                                           Bmat, mask, hw, ATf);

    // 8: per-head O / O3: Ocat[h] = AT[h] (768x768) @ Vcat[h] (768x64, zero-padded)
    wmma_gemm<2,false,false,true><<<gemm_grid(NRES,64,2,NH), blk, 0, stream>>>(
        ATf, VcatT, Ocat, nullptr, NRES, NRES, 64,
        NRES, NRES, 64, NN, (long)64 * NRES, (long)NRES * 64);

    // 9: O_pair: per residue i, (12x768) AT_i^T @ z_i (768x128) -> Ofinal cols 576..2111
    //    (B = z stays row-major: transposing 302 MB would cost more than it saves)
    wmma_gemm<2,true,false,false><<<gemm_grid(12,C_Z,2,NRES), blk, 0, stream>>>(
        ATf, z, Ofinal + 576, nullptr, 12, NRES, C_Z,
        NRES * NRES, C_Z, C_Z, (long)NRES, (long)NRES * C_Z, 2112);

    // 10: O / O3 epilogue into Ofinal cols 0..575
    epilogue_kernel<<<dim3(NRES), dim3(192), 0, stream>>>(Ocat, rots, trn, Ofinal);

    // 11: final projection: out = Ofinal (768x2112) @ Wo (2112x384) + bo
    wmma_gemm<2,false,false,true><<<gemm_grid(NRES,C_S,2,1), blk, 0, stream>>>(
        Ofinal, WoT, out, bo, NRES, 2112, C_S, 2112, 2112, C_S, 0, 0, 0);

    (void)in_sizes; (void)n_in; (void)out_size; (void)ws_size;
}
